// NodeModel_23854248362703
// MI455X (gfx1250) — matrix operand order repaired and checked by reading, not compiled, not run on hardware
//
#include <hip/hip_runtime.h>
#include <hip/hip_bf16.h>

// CDNA5 / gfx1250: wave32, WMMA 16x16x32 bf16 (f32 accumulate).
//
// Strategy: D = A x B with A = W^T tile (16 outF x 32 K, bf16, pre-packed into
// a contiguous-K (K = 16g + j) A-fragment VGPR layout), B = activations (32 K x 16 edges).
// The WMMA C/D layout (lane = column N, VGPRs = rows M) matches the B-operand
// layout (lane = column N, VGPRs = K pairs) up to a lane-group half swap,
// which we do with one __shfl_xor(...,16) per register -> no LDS transpose
// between MLP layers.

typedef __attribute__((ext_vector_type(16))) __bf16 v16bf;
typedef __attribute__((ext_vector_type(8)))  float  v8f;

#define N_NODES 50000
#define N_EDGES 800000
#define NODE_F  128
#define EDGE_F  64
#define HID     128

__device__ __forceinline__ v8f wmma_bf16(v16bf a, v16bf b, v8f c) {
  // 8 args: (neg_a, A, neg_b, B, c_mod, C, reuse_a, reuse_b)
  return __builtin_amdgcn_wmma_f32_16x16x32_bf16(false, a, false, b, (short)0, c,
                                                 false, false);
}

__device__ __forceinline__ v8f zero8() {
  v8f z;
#pragma unroll
  for (int i = 0; i < 8; ++i) z[i] = 0.0f;
  return z;
}

// Pack 16 contiguous f32 (16B aligned) into a bf16 B-fragment half for one lane.
__device__ __forceinline__ v16bf pack16(const float* __restrict__ f) {
  const float4* p = (const float4*)f;
  float4 q0 = p[0], q1 = p[1], q2 = p[2], q3 = p[3];
  float s[16] = {q0.x,q0.y,q0.z,q0.w, q1.x,q1.y,q1.z,q1.w,
                 q2.x,q2.y,q2.z,q2.w, q3.x,q3.y,q3.z,q3.w};
  v16bf r;
#pragma unroll
  for (int i = 0; i < 16; ++i) r[i] = (__bf16)s[i];
  return r;
}

// acc register r holds outF = 16t + 8g + r; bias for 8 consecutive outF.
__device__ __forceinline__ v8f bias_act(v8f a, const float* __restrict__ bp, bool relu) {
  float4 u0 = ((const float4*)bp)[0];
  float4 u1 = ((const float4*)bp)[1];
  float bb[8] = {u0.x,u0.y,u0.z,u0.w, u1.x,u1.y,u1.z,u1.w};
#pragma unroll
  for (int i = 0; i < 8; ++i) {
    float v = a[i] + bb[i];
    a[i] = relu ? fmaxf(v, 0.0f) : v;
  }
  return a;
}

// Convert D tiles (2c, 2c+1) -> next layer B fragment for K-chunk c.
// Lane-group half swap via shfl_xor 16 (wave32).
__device__ __forceinline__ v16bf make_bfrag(const v8f& d0, const v8f& d1, int g) {
  v16bf r;
#pragma unroll
  for (int j = 0; j < 8; ++j) {
    float w  = g ? d0[j] : d1[j];      // what the partner lane-group needs
    float cr = __shfl_xor(w, 16, 32);  // value from lane ^ 16
    float lo = g ? cr    : d0[j];      // K = 32c + 16g + j
    float hi = g ? d1[j] : cr;         // K = 32c + 16g + 8 + j
    r[j]     = (__bf16)lo;
    r[j + 8] = (__bf16)hi;
  }
  return r;
}

// ---------------- weight packing: W[fan_in,128] f32 -> A-fragment bf16 ----------
// A-frag layout (16-bit A 16x32): lane L (l=L&15,g=L>>4): half j holds
// K = 32c + 16g + j, M(outF) = 16t + l.  Frag (c,t) at (c*8+t)*512 halfs.
__global__ void pack_w(const float* __restrict__ W, __bf16* __restrict__ dst, int nChunks) {
  int tid  = blockIdx.x * blockDim.x + threadIdx.x;
  int frag = tid >> 5, lane = tid & 31;
  if (frag >= nChunks * 8) return;
  int c = frag >> 3, t = frag & 7;
  int g = lane >> 4, l = lane & 15;
  __bf16* o = dst + (size_t)frag * 512 + lane * 16;
#pragma unroll
  for (int j = 0; j < 16; ++j) {
    int K = 32 * c + 16 * g + j;
    o[j] = (__bf16)W[(size_t)K * 128 + 16 * t + l];
  }
}

__global__ void zero_f32(float* __restrict__ p, int n) {
  for (int i = blockIdx.x * blockDim.x + threadIdx.x; i < n; i += gridDim.x * blockDim.x)
    p[i] = 0.0f;
}

// One K-chunk-set GEMM over 8 outF tiles, two 16-column tiles (A/B edge tiles).
template <int NC>
__device__ __forceinline__ void gemm2(const v16bf* fA, const v16bf* fB,
                                      const __bf16* __restrict__ wp, int lane,
                                      v8f* accA, v8f* accB) {
#pragma unroll
  for (int c = 0; c < NC; ++c) {
    const __bf16* wl = wp + (size_t)c * 8 * 512 + lane * 16;
#pragma unroll
    for (int t = 0; t < 8; ++t) {
      v16bf af = *(const v16bf*)(wl + t * 512);
      accA[t] = wmma_bf16(af, fA[c], accA[t]);
      accB[t] = wmma_bf16(af, fB[c], accB[t]);
    }
  }
}

template <int NC>
__device__ __forceinline__ void gemm1(const v16bf* f, const __bf16* __restrict__ wp,
                                      int lane, v8f* acc) {
#pragma unroll
  for (int c = 0; c < NC; ++c) {
    const __bf16* wl = wp + (size_t)c * 8 * 512 + lane * 16;
#pragma unroll
    for (int t = 0; t < 8; ++t) {
      v16bf af = *(const v16bf*)(wl + t * 512);
      acc[t] = wmma_bf16(af, f[c], acc[t]);
    }
  }
}

// ---------------- edge MLP + scatter-add -------------------------------------
__global__ __launch_bounds__(256) void edge_mlp(
    const float* __restrict__ x, const int* __restrict__ ei,
    const float* __restrict__ ea,
    const __bf16* __restrict__ w0p, const float* __restrict__ b0,
    const __bf16* __restrict__ w1p, const float* __restrict__ b1,
    const __bf16* __restrict__ w2p, const float* __restrict__ b2,
    float* __restrict__ agg) {
  const int lane = threadIdx.x & 31;
  const int wv   = (blockIdx.x * blockDim.x + threadIdx.x) >> 5;
  const int g = lane >> 4, l = lane & 15;
  const int e0 = wv * 32;                 // two 16-edge tiles per wave
  const int eA = e0 + l, eB = e0 + 16 + l;
  const int rowA = ei[eA], rowB = ei[eB];
  const int colA = ei[N_EDGES + eA], colB = ei[N_EDGES + eB];

  v8f accA[8], accB[8];
#pragma unroll
  for (int t = 0; t < 8; ++t) { accA[t] = zero8(); accB[t] = zero8(); }

  // ---- layer 0: K = 192 (x[row] 0..127 | edge_attr 128..191), f32->bf16 on the fly
#pragma unroll
  for (int c = 0; c < 6; ++c) {
    const float* sA = (c < 4) ? (x + (size_t)rowA * NODE_F + 32 * c + 16 * g)
                              : (ea + (size_t)eA * EDGE_F + (32 * c - NODE_F) + 16 * g);
    const float* sB = (c < 4) ? (x + (size_t)rowB * NODE_F + 32 * c + 16 * g)
                              : (ea + (size_t)eB * EDGE_F + (32 * c - NODE_F) + 16 * g);
    v16bf bA = pack16(sA), bB = pack16(sB);
    const __bf16* wl = w0p + (size_t)c * 8 * 512 + lane * 16;
#pragma unroll
    for (int t = 0; t < 8; ++t) {
      v16bf af = *(const v16bf*)(wl + t * 512);
      accA[t] = wmma_bf16(af, bA, accA[t]);
      accB[t] = wmma_bf16(af, bB, accB[t]);
    }
  }
#pragma unroll
  for (int t = 0; t < 8; ++t) {
    accA[t] = bias_act(accA[t], b0 + 16 * t + 8 * g, true);
    accB[t] = bias_act(accB[t], b0 + 16 * t + 8 * g, true);
  }

  // ---- layer 1
  v16bf fA[4], fB[4];
#pragma unroll
  for (int c = 0; c < 4; ++c) {
    fA[c] = make_bfrag(accA[2 * c], accA[2 * c + 1], g);
    fB[c] = make_bfrag(accB[2 * c], accB[2 * c + 1], g);
  }
#pragma unroll
  for (int t = 0; t < 8; ++t) { accA[t] = zero8(); accB[t] = zero8(); }
  gemm2<4>(fA, fB, w1p, lane, accA, accB);
#pragma unroll
  for (int t = 0; t < 8; ++t) {
    accA[t] = bias_act(accA[t], b1 + 16 * t + 8 * g, true);
    accB[t] = bias_act(accB[t], b1 + 16 * t + 8 * g, true);
  }

  // ---- layer 2 (no relu)
#pragma unroll
  for (int c = 0; c < 4; ++c) {
    fA[c] = make_bfrag(accA[2 * c], accA[2 * c + 1], g);
    fB[c] = make_bfrag(accB[2 * c], accB[2 * c + 1], g);
  }
#pragma unroll
  for (int t = 0; t < 8; ++t) { accA[t] = zero8(); accB[t] = zero8(); }
  gemm2<4>(fA, fB, w2p, lane, accA, accB);
#pragma unroll
  for (int t = 0; t < 8; ++t) {
    accA[t] = bias_act(accA[t], b2 + 16 * t + 8 * g, false);
    accB[t] = bias_act(accB[t], b2 + 16 * t + 8 * g, false);
  }

  // ---- scatter-add (segment_sum): L2-resident f32 atomics
#pragma unroll
  for (int t = 0; t < 8; ++t) {
    const int fb = 16 * t + 8 * g;
#pragma unroll
    for (int r = 0; r < 8; ++r) {
      atomicAdd(agg + (size_t)colA * HID + fb + r, accA[t][r]);
      atomicAdd(agg + (size_t)colB * HID + fb + r, accB[t][r]);
    }
  }
}

// ---------------- node MLP ---------------------------------------------------
__global__ __launch_bounds__(256) void node_mlp(
    const float* __restrict__ x, const float* __restrict__ agg,
    const __bf16* __restrict__ v0p, const float* __restrict__ c0,
    const __bf16* __restrict__ v1p, const float* __restrict__ c1,
    const __bf16* __restrict__ v2p, const float* __restrict__ c2,
    float* __restrict__ out) {
  const int lane = threadIdx.x & 31;
  const int wv   = (blockIdx.x * blockDim.x + threadIdx.x) >> 5;
  if (wv * 16 >= N_NODES) return;  // wave-uniform exit (EXEC stays full for WMMA)
  const int g = lane >> 4, l = lane & 15;
  const int node = wv * 16 + l;

  v8f acc[8];
#pragma unroll
  for (int t = 0; t < 8; ++t) acc[t] = zero8();

  // ---- layer 0: K = 256 (x 0..127 | agg 128..255)
#pragma unroll
  for (int c = 0; c < 8; ++c) {
    const float* s = (c < 4) ? (x + (size_t)node * NODE_F + 32 * c + 16 * g)
                             : (agg + (size_t)node * HID + 32 * (c - 4) + 16 * g);
    v16bf bf = pack16(s);
    const __bf16* wl = v0p + (size_t)c * 8 * 512 + lane * 16;
#pragma unroll
    for (int t = 0; t < 8; ++t) {
      v16bf af = *(const v16bf*)(wl + t * 512);
      acc[t] = wmma_bf16(af, bf, acc[t]);
    }
  }
#pragma unroll
  for (int t = 0; t < 8; ++t) acc[t] = bias_act(acc[t], c0 + 16 * t + 8 * g, true);

  v16bf f[4];
#pragma unroll
  for (int c = 0; c < 4; ++c) f[c] = make_bfrag(acc[2 * c], acc[2 * c + 1], g);
#pragma unroll
  for (int t = 0; t < 8; ++t) acc[t] = zero8();
  gemm1<4>(f, v1p, lane, acc);
#pragma unroll
  for (int t = 0; t < 8; ++t) acc[t] = bias_act(acc[t], c1 + 16 * t + 8 * g, true);

#pragma unroll
  for (int c = 0; c < 4; ++c) f[c] = make_bfrag(acc[2 * c], acc[2 * c + 1], g);
#pragma unroll
  for (int t = 0; t < 8; ++t) acc[t] = zero8();
  gemm1<4>(f, v2p, lane, acc);
#pragma unroll
  for (int t = 0; t < 8; ++t) acc[t] = bias_act(acc[t], c2 + 16 * t + 8 * g, false);

  // ---- store: 8 contiguous outF per register set -> two float4 stores per tile
#pragma unroll
  for (int t = 0; t < 8; ++t) {
    const int fb = 16 * t + 8 * g;
    float4 o0 = make_float4(acc[t][0], acc[t][1], acc[t][2], acc[t][3]);
    float4 o1 = make_float4(acc[t][4], acc[t][5], acc[t][6], acc[t][7]);
    float4* dst = (float4*)(out + (size_t)node * NODE_F + fb);
    dst[0] = o0;
    dst[1] = o1;
  }
}

extern "C" void kernel_launch(void* const* d_in, const int* in_sizes, int n_in,
                              void* d_out, int out_size, void* d_ws, size_t ws_size,
                              hipStream_t stream) {
  const float* x  = (const float*)d_in[0];
  const int*   ei = (const int*)d_in[1];
  const float* ea = (const float*)d_in[2];
  // d_in[3] = u, d_in[4] = batch: unused by the reference math
  const float* W0 = (const float*)d_in[5];  const float* b0 = (const float*)d_in[6];
  const float* W1 = (const float*)d_in[7];  const float* b1 = (const float*)d_in[8];
  const float* W2 = (const float*)d_in[9];  const float* b2 = (const float*)d_in[10];
  const float* V0 = (const float*)d_in[11]; const float* c0 = (const float*)d_in[12];
  const float* V1 = (const float*)d_in[13]; const float* c1 = (const float*)d_in[14];
  const float* V2 = (const float*)d_in[15]; const float* c2 = (const float*)d_in[16];

  // Workspace layout (~24.7 MiB total)
  char*  ws  = (char*)d_ws;
  float* agg = (float*)ws;
  size_t off = (size_t)N_NODES * HID * sizeof(float);       // 25,600,000 B (32B aligned)
  __bf16* w0p = (__bf16*)(ws + off); off += 48 * 1024;      // 6 chunks * 8 tiles
  __bf16* w1p = (__bf16*)(ws + off); off += 32 * 1024;      // 4 * 8
  __bf16* w2p = (__bf16*)(ws + off); off += 32 * 1024;
  __bf16* v0p = (__bf16*)(ws + off); off += 64 * 1024;      // 8 * 8
  __bf16* v1p = (__bf16*)(ws + off); off += 32 * 1024;
  __bf16* v2p = (__bf16*)(ws + off); off += 32 * 1024;

  // 1) pack weights into WMMA A-fragment layout (1 block per K-chunk: 8 frags * 32 lanes)
  pack_w<<<6, 256, 0, stream>>>(W0, w0p, 6);
  pack_w<<<4, 256, 0, stream>>>(W1, w1p, 4);
  pack_w<<<4, 256, 0, stream>>>(W2, w2p, 4);
  pack_w<<<8, 256, 0, stream>>>(V0, v0p, 8);
  pack_w<<<4, 256, 0, stream>>>(V1, v1p, 4);
  pack_w<<<4, 256, 0, stream>>>(V2, v2p, 4);

  // 2) zero the scatter accumulator (every launch: atomics accumulate into it)
  zero_f32<<<2048, 256, 0, stream>>>(agg, N_NODES * HID);

  // 3) edge MLP + scatter-add: 32 edges/wave, 8 waves/block -> 800000/256 blocks
  edge_mlp<<<N_EDGES / 256, 256, 0, stream>>>(x, ei, ea, w0p, b0, w1p, b1, w2p, b2, agg);

  // 4) node MLP: 16 nodes/wave -> 3125 waves -> 391 blocks
  node_mlp<<<(N_NODES / 16 + 7) / 8, 256, 0, stream>>>(x, agg, v0p, c0, v1p, c1, v2p, c2,
                                                       (float*)d_out);
}
